// CausalSelfAttention_34333968564596
// MI455X (gfx1250) — matrix operand order repaired and checked by reading, not compiled, not run on hardware
//
#include <hip/hip_runtime.h>

// ---------------------------------------------------------------------------
// CDNA5 (gfx1250) causal self-attention: bf16 WMMA + async global->LDS DMA.
//   x[4096,768] @ Wqkv[768,2304] + bqkv -> qkv (bf16)
//   flash attention (causal, online softmax, 12 heads, hd=64) -> O (bf16)
//   O @ Wout[768,768] + bout -> out (f32)
// Data movement: GLOBAL_LOAD_ASYNC_TO_LDS_B128 with double-buffered LDS,
// synchronized via ASYNCcnt (s_wait_asynccnt) + workgroup barriers.
// ---------------------------------------------------------------------------

typedef unsigned short u16;
typedef __bf16 bf16_t;
typedef bf16_t v16bf __attribute__((ext_vector_type(16)));
typedef float  v8f   __attribute__((ext_vector_type(8)));

#define S_LEN   4096
#define D_MODEL 768
#define NHEAD   12
#define HD      64
#define QKV_LD  2304

struct alignas(16) U128 { unsigned int x, y, z, w; };
struct B32x2  { U128 lo, hi; };   // 32 bytes -> one v16bf
struct U16x16 { u16 s[16]; };     // 32 bytes -> one v16bf

using lds_void = __attribute__((address_space(3))) void;

// async DMA: 16B per lane, global -> LDS, tracked by ASYNCcnt
__device__ __forceinline__ void async_copy_b128(void* lds_dst,
                                                const void* gsrc) {
  asm volatile("global_load_async_to_lds_b128 %0, %1, off"
               :
               : "v"((lds_void*)lds_dst), "v"(gsrc)
               : "memory");
}

#if __has_builtin(__builtin_amdgcn_s_wait_asynccnt)
#define WAIT_ASYNCCNT(n) __builtin_amdgcn_s_wait_asynccnt(n)
#else
#define WAIT_ASYNCCNT(n) asm volatile("s_wait_asynccnt %0" ::"i"(n) : "memory")
#endif

__device__ __forceinline__ u16 f2bf_rne(float f) {
  union { float f; unsigned u; } v; v.f = f;
  unsigned u = v.u;
  u += 0x7fffu + ((u >> 16) & 1u);   // round-to-nearest-even
  return (u16)(u >> 16);
}

__device__ __forceinline__ v8f v8f_zero() {
  v8f z;
#pragma unroll
  for (int i = 0; i < 8; ++i) z[i] = 0.0f;
  return z;
}

// ---------------------------------------------------------------------------
// elementwise f32 -> bf16
// ---------------------------------------------------------------------------
__global__ void f32_to_bf16_kernel(const float* __restrict__ in,
                                   u16* __restrict__ out, int n) {
  int i = blockIdx.x * blockDim.x + threadIdx.x;
  int stride = gridDim.x * blockDim.x;
  for (; i < n; i += stride) out[i] = f2bf_rne(in[i]);
}

// ---------------------------------------------------------------------------
// Tiled GEMM:  C[M,N] = A[M,K](bf16) * B[K,N](bf16) + bias[N]
// BM=128 BN=64 BK=32, 256 threads = 8 waves, each wave a 32x32 tile.
// Double-buffered LDS, async-DMA staged (3 async ops / thread / stage).
// ---------------------------------------------------------------------------
template <bool F32OUT>
__global__ __launch_bounds__(256) void gemm_bf16_kernel(
    const u16* __restrict__ A, const u16* __restrict__ B,
    const float* __restrict__ bias, void* __restrict__ Cout,
    int M, int N, int K) {
  __shared__ u16 As[2][128][40];   // 32 cols + 8 pad (row 80B, 16B aligned)
  __shared__ u16 Bs[2][32][72];    // 64 cols + 8 pad (row 144B, 16B aligned)

  const int tid  = threadIdx.x;
  const int w    = tid >> 5;
  const int lane = tid & 31;
  const int h    = lane >> 4;     // lane half
  const int nl   = lane & 15;
  const int wm   = (w >> 1) * 32;
  const int wn   = (w & 1) * 32;
  const int m0   = blockIdx.y * 128;
  const int n0   = blockIdx.x * 64;

  const int ar = tid >> 2, ac = (tid & 3) * 8;   // A slots: rows ar, ar+64
  const int br = tid >> 3, bc = (tid & 7) * 8;   // B slot

  auto issue_stage = [&](int buf, int k0) {
    async_copy_b128(&As[buf][ar][ac],
                    &A[(size_t)(m0 + ar) * K + k0 + ac]);
    async_copy_b128(&As[buf][ar + 64][ac],
                    &A[(size_t)(m0 + ar + 64) * K + k0 + ac]);
    async_copy_b128(&Bs[buf][br][bc],
                    &B[(size_t)(k0 + br) * N + n0 + bc]);
  };

  v8f acc[2][2];
#pragma unroll
  for (int ti = 0; ti < 2; ++ti)
#pragma unroll
    for (int tj = 0; tj < 2; ++tj) acc[ti][tj] = v8f_zero();

  const int nk = K >> 5;
  issue_stage(0, 0);

  for (int ik = 0; ik < nk; ++ik) {
    const int cur = ik & 1;
    if (ik + 1 < nk) {
      issue_stage(cur ^ 1, (ik + 1) << 5);  // prefetch next tile
      WAIT_ASYNCCNT(3);                     // current stage retired
    } else {
      WAIT_ASYNCCNT(0);
    }
    __syncthreads();

    // A fragments (lane row = nl ; k = 16h + e, contiguous)
    v16bf afrag[2];
#pragma unroll
    for (int ti = 0; ti < 2; ++ti) {
      const U128* p = (const U128*)&As[cur][wm + ti * 16 + nl][h * 16];
      B32x2 t; t.lo = p[0]; t.hi = p[1];
      afrag[ti] = __builtin_bit_cast(v16bf, t);
    }
    // B fragments (lane col = nl ; k = 16h + e) — strided column gather
    v16bf bfrag[2];
#pragma unroll
    for (int tj = 0; tj < 2; ++tj) {
      U16x16 t;
#pragma unroll
      for (int e = 0; e < 16; ++e)
        t.s[e] = Bs[cur][16 * h + e][wn + tj * 16 + nl];
      bfrag[tj] = __builtin_bit_cast(v16bf, t);
    }
#pragma unroll
    for (int ti = 0; ti < 2; ++ti)
#pragma unroll
      for (int tj = 0; tj < 2; ++tj)
        acc[ti][tj] = __builtin_amdgcn_wmma_f32_16x16x32_bf16(
            false, afrag[ti], false, bfrag[tj], (short)0, acc[ti][tj],
            false, false);
    __syncthreads();  // all reads of buf `cur` done before it is re-filled
  }

  // epilogue: C layout — lane = col, VGPR r = row (r + 8*half)
#pragma unroll
  for (int ti = 0; ti < 2; ++ti)
#pragma unroll
    for (int tj = 0; tj < 2; ++tj) {
      const int gn = n0 + wn + tj * 16 + nl;
      const float bv = bias[gn];
#pragma unroll
      for (int r = 0; r < 8; ++r) {
        const int gm = m0 + wm + ti * 16 + 8 * h + r;
        const float v = acc[ti][tj][r] + bv;
        if (F32OUT)
          ((float*)Cout)[(size_t)gm * N + gn] = v;
        else
          ((u16*)Cout)[(size_t)gm * N + gn] = f2bf_rne(v);
      }
    }
}

// ---------------------------------------------------------------------------
// Flash attention, causal. grid = (S/128, NHEAD), 256 threads (8 waves),
// each wave owns 16 query rows; 64-key K/V blocks double-buffered in LDS
// via async DMA (4 async ops / thread / stage).
// ---------------------------------------------------------------------------
__global__ __launch_bounds__(256) void flash_attn_kernel(
    const u16* __restrict__ qkv, u16* __restrict__ O) {
  __shared__ u16 Qs[128][72];
  __shared__ u16 Ks[2][64][72];
  __shared__ u16 Vs[2][64][72];
  __shared__ u16 Ps[8][16][72];   // per-wave P staging (C-layout -> A-layout)

  const int tid  = threadIdx.x;
  const int w    = tid >> 5;
  const int lane = tid & 31;
  const int h    = lane >> 4;
  const int nl   = lane & 15;
  const int qb   = blockIdx.x;
  const int head = blockIdx.y;
  const int q0   = qb * 128;
  const size_t qoff = (size_t)head * HD;
  const size_t koff = (size_t)D_MODEL + (size_t)head * HD;
  const size_t voff = (size_t)(2 * D_MODEL) + (size_t)head * HD;

  auto issue_kv = [&](int buf, int kb) {
#pragma unroll
    for (int i = 0; i < 2; ++i) {
      int slot = tid + i * 256;
      int r = slot >> 3;
      int c = (slot & 7) * 8;
      async_copy_b128(&Ks[buf][r][c],
                      &qkv[(size_t)(kb * 64 + r) * QKV_LD + koff + c]);
      async_copy_b128(&Vs[buf][r][c],
                      &qkv[(size_t)(kb * 64 + r) * QKV_LD + voff + c]);
    }
  };

  // ---- async-load Q tile 128x64 (4 ops), then stage 0 of K/V (4 ops)
#pragma unroll
  for (int i = 0; i < 4; ++i) {
    int slot = tid + i * 256;
    int r = slot >> 3;
    int c = (slot & 7) * 8;
    async_copy_b128(&Qs[r][c], &qkv[(size_t)(q0 + r) * QKV_LD + qoff + c]);
  }
  issue_kv(0, 0);
  WAIT_ASYNCCNT(4);   // Q's 4 ops retired (in-order); K/V stage 0 in flight
  __syncthreads();

  // Q as two A-fragments (K chunks 0..31, 32..63)
  v16bf qa[2];
#pragma unroll
  for (int kc = 0; kc < 2; ++kc) {
    const U128* p = (const U128*)&Qs[w * 16 + nl][kc * 32 + h * 16];
    B32x2 t; t.lo = p[0]; t.hi = p[1];
    qa[kc] = __builtin_bit_cast(v16bf, t);
  }

  v8f o_acc[4];
#pragma unroll
  for (int j = 0; j < 4; ++j) o_acc[j] = v8f_zero();
  float m_i[8], l_i[8];
#pragma unroll
  for (int r = 0; r < 8; ++r) { m_i[r] = -1.0e30f; l_i[r] = 0.0f; }

  const float ssc = 0.125f * 1.44269504088896340736f;  // 1/sqrt(64)*log2(e)

  const int nkb = qb * 2 + 2;  // key blocks up to (and including) diagonal
  for (int kb = 0; kb < nkb; ++kb) {
    const int cur = kb & 1;
    if (kb + 1 < nkb) {
      issue_kv(cur ^ 1, kb + 1);  // prefetch next key block
      WAIT_ASYNCCNT(4);
    } else {
      WAIT_ASYNCCNT(0);
    }
    __syncthreads();

    // ---- preload all 8 K^T B-fragments, then 8 back-to-back WMMAs
    v16bf kfrag[4][2];
#pragma unroll
    for (int j = 0; j < 4; ++j)
#pragma unroll
      for (int kc = 0; kc < 2; ++kc) {
        // B(k,n) = K[key=j*16+nl][kc*32+k]; k = 16h+e -> 16 contiguous u16
        const U128* p = (const U128*)&Ks[cur][j * 16 + nl][kc * 32 + h * 16];
        B32x2 t; t.lo = p[0]; t.hi = p[1];
        kfrag[j][kc] = __builtin_bit_cast(v16bf, t);
      }
    v8f s[4];
#pragma unroll
    for (int j = 0; j < 4; ++j)
      s[j] = __builtin_amdgcn_wmma_f32_16x16x32_bf16(
          false, qa[0], false, kfrag[j][0], (short)0, v8f_zero(), false,
          false);
#pragma unroll
    for (int j = 0; j < 4; ++j)
      s[j] = __builtin_amdgcn_wmma_f32_16x16x32_bf16(
          false, qa[1], false, kfrag[j][1], (short)0, s[j], false, false);

    // ---- scale + causal mask
    float sc[4][8];
#pragma unroll
    for (int j = 0; j < 4; ++j) {
      const int key_g = kb * 64 + j * 16 + nl;
#pragma unroll
      for (int r = 0; r < 8; ++r) {
        const int q_g = q0 + w * 16 + 8 * h + r;
        sc[j][r] = (key_g > q_g) ? -1.0e30f : s[j][r] * ssc;
      }
    }

    // ---- online softmax (rows = 8h+r; reduce over the 16 lanes of a half)
#pragma unroll
    for (int r = 0; r < 8; ++r) {
      float rm = fmaxf(fmaxf(sc[0][r], sc[1][r]), fmaxf(sc[2][r], sc[3][r]));
#pragma unroll
      for (int msk = 1; msk < 16; msk <<= 1)
        rm = fmaxf(rm, __shfl_xor(rm, msk, 32));
      const float mn = fmaxf(m_i[r], rm);
      const float alpha = exp2f(m_i[r] - mn);
      m_i[r] = mn;
      float rs = 0.0f;
#pragma unroll
      for (int j = 0; j < 4; ++j) {
        const float p = exp2f(sc[j][r] - mn);
        sc[j][r] = p;
        rs += p;
      }
#pragma unroll
      for (int msk = 1; msk < 16; msk <<= 1)
        rs += __shfl_xor(rs, msk, 32);
      l_i[r] = l_i[r] * alpha + rs;
#pragma unroll
      for (int j = 0; j < 4; ++j) o_acc[j][r] *= alpha;
    }

    // ---- P: C-layout -> LDS (bf16) -> A-layout fragments
#pragma unroll
    for (int j = 0; j < 4; ++j)
#pragma unroll
      for (int r = 0; r < 8; ++r)
        Ps[w][8 * h + r][j * 16 + nl] = f2bf_rne(sc[j][r]);
    // (same-wave LDS ordering: stores precede loads in program order)

    // ---- preload P A-fragments and all 8 V B-fragments, then 8 WMMAs
    v16bf pa[2];
#pragma unroll
    for (int kc = 0; kc < 2; ++kc) {
      const U128* p = (const U128*)&Ps[w][nl][kc * 32 + h * 16];
      B32x2 t; t.lo = p[0]; t.hi = p[1];
      pa[kc] = __builtin_bit_cast(v16bf, t);
    }
    v16bf vfrag[4][2];
#pragma unroll
    for (int j = 0; j < 4; ++j)
#pragma unroll
      for (int kc = 0; kc < 2; ++kc) {
        U16x16 t;
#pragma unroll
        for (int e = 0; e < 16; ++e)
          t.s[e] = Vs[cur][kc * 32 + 16 * h + e][j * 16 + nl];
        vfrag[j][kc] = __builtin_bit_cast(v16bf, t);
      }
#pragma unroll
    for (int kc = 0; kc < 2; ++kc)
#pragma unroll
      for (int j = 0; j < 4; ++j)
        o_acc[j] = __builtin_amdgcn_wmma_f32_16x16x32_bf16(
            false, pa[kc], false, vfrag[j][kc], (short)0, o_acc[j], false,
            false);

    __syncthreads();  // buf `cur` fully consumed before re-fill
  }

  // ---- finalize: O /= l, store bf16 [S, D] head-sliced
#pragma unroll
  for (int r = 0; r < 8; ++r) {
    const float inv = 1.0f / l_i[r];
    const int gm = q0 + w * 16 + 8 * h + r;
#pragma unroll
    for (int j = 0; j < 4; ++j)
      O[(size_t)gm * D_MODEL + head * HD + j * 16 + nl] =
          f2bf_rne(o_acc[j][r] * inv);
  }
}

// ---------------------------------------------------------------------------
// launcher
// ---------------------------------------------------------------------------
extern "C" void kernel_launch(void* const* d_in, const int* in_sizes, int n_in,
                              void* d_out, int out_size, void* d_ws,
                              size_t ws_size, hipStream_t stream) {
  const float* x    = (const float*)d_in[0];
  const float* Wqkv = (const float*)d_in[1];
  const float* bqkv = (const float*)d_in[2];
  const float* Wout = (const float*)d_in[3];
  const float* bout = (const float*)d_in[4];
  float* out = (float*)d_out;

  char* ws = (char*)d_ws;
  u16* x_bf    = (u16*)(ws + 0);         // 4096*768*2  = 6291456
  u16* wqkv_bf = (u16*)(ws + 6291456);   // 768*2304*2  = 3538944
  u16* wout_bf = (u16*)(ws + 9830400);   // 768*768*2   = 1179648
  u16* qkv_bf  = (u16*)(ws + 11010048);  // 4096*2304*2 = 18874368
  u16* attn_bf = (u16*)(ws + 29884416);  // 4096*768*2  = 6291456

  f32_to_bf16_kernel<<<1024, 256, 0, stream>>>(x, x_bf, S_LEN * D_MODEL);
  f32_to_bf16_kernel<<<1024, 256, 0, stream>>>(Wqkv, wqkv_bf,
                                               D_MODEL * QKV_LD);
  f32_to_bf16_kernel<<<512, 256, 0, stream>>>(Wout, wout_bf,
                                              D_MODEL * D_MODEL);

  gemm_bf16_kernel<false><<<dim3(QKV_LD / 64, S_LEN / 128), 256, 0, stream>>>(
      x_bf, wqkv_bf, bqkv, qkv_bf, S_LEN, QKV_LD, D_MODEL);

  flash_attn_kernel<<<dim3(S_LEN / 128, NHEAD), 256, 0, stream>>>(qkv_bf,
                                                                  attn_bf);

  gemm_bf16_kernel<true><<<dim3(D_MODEL / 64, S_LEN / 128), 256, 0, stream>>>(
      attn_bf, wout_bf, bout, out, S_LEN, D_MODEL, D_MODEL);
}
